// MoE_1563368095780
// MI455X (gfx1250) — matrix operand order repaired and checked by reading, not compiled, not run on hardware
//
#include <hip/hip_runtime.h>
#include <hip/hip_bf16.h>
#include <cstdint>

// Problem constants (from reference)
#define BDIM 2
#define SDIM 2048
#define TTOK (BDIM * SDIM) // 4096 tokens
#define DDIM 1024
#define HDIM 2048
#define EEXP 8
#define KTOP 2

// LDS strides (in uint = 2 bf16), padded to stagger banks
#define XS 520   // x tile row stride (512 data + 8 pad)  -> 2080 B
#define HS 1032  // h tile row stride (1024 data + 8 pad) -> 4128 B

typedef __attribute__((ext_vector_type(16))) __bf16 bf16x16;
typedef __attribute__((ext_vector_type(8)))  float  v8f;
typedef unsigned int u32x4 __attribute__((ext_vector_type(4)));
typedef int          i32x4 __attribute__((ext_vector_type(4)));
typedef int          i32x8 __attribute__((ext_vector_type(8)));

union BOp { unsigned int u[8]; bf16x16 v; };

__device__ __forceinline__ unsigned int pk_bf16(float lo, float hi) {
    unsigned int a = __float_as_uint(lo);
    unsigned int b = __float_as_uint(hi);
    a = (a + 0x7FFFu + ((a >> 16) & 1u)) >> 16;   // round-to-nearest-even
    b = (b + 0x7FFFu + ((b >> 16) & 1u)) >> 16;
    return (b << 16) | (a & 0xFFFFu);
}

__device__ __forceinline__ unsigned short bf16_1(float x) {
    unsigned int a = __float_as_uint(x);
    return (unsigned short)((a + 0x7FFFu + ((a >> 16) & 1u)) >> 16);
}

// ---------------------------------------------------------------------------
// Kernel 0: f32 -> bf16 conversion prepass (pair-packed, coalesced)
// ---------------------------------------------------------------------------
__global__ void __launch_bounds__(256)
cvt_f32_bf16(const float* __restrict__ src, unsigned int* __restrict__ dst,
             long npairs) {
    long i = (long)blockIdx.x * blockDim.x + threadIdx.x;
    if (i >= npairs) return;
    const float2* s2 = (const float2*)src;
    float2 f = s2[i];
    dst[i] = pk_bf16(f.x, f.y);
}

// ---------------------------------------------------------------------------
// Kernel 1: router. One wave32 per token.
// ---------------------------------------------------------------------------
__global__ void __launch_bounds__(256)
moe_router(const float* __restrict__ x,
           const float* __restrict__ Wr,
           const float* __restrict__ br,
           float* __restrict__ gate_logits_out,  // [T,E]
           float* __restrict__ topk_out,         // [T,K] (indices as floats)
           float* __restrict__ gws)              // [T,E] sparse gate weights
{
    const int t    = (blockIdx.x * blockDim.x + threadIdx.x) >> 5;
    const int lane = threadIdx.x & 31;
    if (t >= TTOK) return;

    const float* xt = x + (size_t)t * DDIM;
    float acc[EEXP];
#pragma unroll
    for (int e = 0; e < EEXP; ++e) acc[e] = 0.f;

    for (int d = lane; d < DDIM; d += 32) {
        float xv = xt[d];
#pragma unroll
        for (int e = 0; e < EEXP; ++e)
            acc[e] += xv * Wr[e * DDIM + d];
    }
#pragma unroll
    for (int e = 0; e < EEXP; ++e) {
#pragma unroll
        for (int off = 16; off >= 1; off >>= 1)
            acc[e] += __shfl_xor(acc[e], off, 32);
        acc[e] += br[e];
    }

    if (lane == 0) {
        float mx = acc[0];
#pragma unroll
        for (int e = 1; e < EEXP; ++e) mx = fmaxf(mx, acc[e]);
        float p[EEXP];
#pragma unroll
        for (int e = 0; e < EEXP; ++e) p[e] = __expf(acc[e] - mx);
        int i1 = 0; float p1 = p[0];
#pragma unroll
        for (int e = 1; e < EEXP; ++e) if (p[e] > p1) { p1 = p[e]; i1 = e; }
        int i2 = -1; float p2 = -1.f;
#pragma unroll
        for (int e = 0; e < EEXP; ++e)
            if (e != i1 && p[e] > p2) { p2 = p[e]; i2 = e; }
        float inv2 = __builtin_amdgcn_rcpf(p1 + p2);

#pragma unroll
        for (int e = 0; e < EEXP; ++e) {
            gate_logits_out[t * EEXP + e] = acc[e];
            gws[t * EEXP + e] = (e == i1) ? p1 * inv2 : ((e == i2) ? p2 * inv2 : 0.f);
        }
        topk_out[t * KTOP + 0] = (float)i1;
        topk_out[t * KTOP + 1] = (float)i2;
    }
}

// ---------------------------------------------------------------------------
// Kernel 2: deterministic init of out with gate-weighted biases
// ---------------------------------------------------------------------------
__global__ void __launch_bounds__(256)
moe_bias_init(const float* __restrict__ bo,    // [E,D]
              const float* __restrict__ bo_s,  // [D]
              const float* __restrict__ gws,   // [T,E]
              float* __restrict__ out)         // [T,D]
{
    int idx = blockIdx.x * blockDim.x + threadIdx.x;
    if (idx >= TTOK * DDIM) return;
    int t = idx >> 10;
    int d = idx & (DDIM - 1);
    float v = bo_s[d];
    const float* g = gws + t * EEXP;
#pragma unroll
    for (int e = 0; e < EEXP; ++e) v += g[e] * bo[e * DDIM + d];
    out[idx] = v;
}

// ---------------------------------------------------------------------------
// TDM helper: load 16 rows x 2048B of bf16 x-tile into LDS (iterate mode,
// LDS row stride XS*4 bytes). Falls back to plain copy if builtin missing.
// ---------------------------------------------------------------------------
__device__ __forceinline__ void tdm_load_x_tile(const unsigned short* gsrc,
                                                unsigned int* sx, int tid) {
#if __has_builtin(__builtin_amdgcn_tensor_load_to_lds)
    if (tid < 32) {   // wave 0 issues one TDM op; TENSORcnt is per-wave
        unsigned long long ga = (unsigned long long)(uintptr_t)gsrc;
        unsigned int la = (unsigned int)(uintptr_t)sx;   // LDS aperture low bits

        u32x4 g0 = { 1u,                                   // count=1, user desc
                     la,                                    // lds_addr
                     (unsigned int)(ga & 0xFFFFFFFFu),      // global_addr[31:0]
                     (unsigned int)((ga >> 32) & 0x01FFFFFFu) | (2u << 30) }; // +type=2

        // data_size=1 (2B), iterate_enable=1; tensor_dim0=1024, tensor_dim1=4096
        // tile_dim0=1024, tile_dim1=1; tensor_dim0_stride=1024
        i32x8 g1 = { (int)((1u << 16) | (1u << 19)),
                     (int)(1024u << 16),
                     (int)(4096u << 16),
                     (int)(1024u << 16),
                     1,
                     1024,
                     0, 0 };
        // iterate: lds_addr_increment=XS*2 (=1040 elems), global_addr_increment=1024,
        // iterate_count=15 (16 rows)
        i32x4 g2 = { 0, (int)(XS * 2), 1024, (int)(15u << 16) };
        i32x4 g3 = { 0, 0, 0, 0 };
#if defined(__clang_major__) && (__clang_major__ >= 23)
        i32x8 g4 = { 0, 0, 0, 0, 0, 0, 0, 0 };
        __builtin_amdgcn_tensor_load_to_lds(g0, g1, g2, g3, g4, 0);
#else
        __builtin_amdgcn_tensor_load_to_lds(g0, g1, g2, g3, 0);
#endif
#if __has_builtin(__builtin_amdgcn_s_wait_tensorcnt)
        __builtin_amdgcn_s_wait_tensorcnt(0);
#endif
    }
#else
    // Fallback: cooperative copy of 16 x 512 uints (bf16 pairs)
    const uint4* g4 = (const uint4*)gsrc;
    for (int i = tid; i < 16 * 128; i += 128) {
        int r = i >> 7, c = i & 127;           // c indexes uint4 within row
        uint4 v = g4[r * 128 + c];
        uint4* d = (uint4*)&sx[r * XS + c * 4];
        *d = v;
    }
#endif
}

// ---------------------------------------------------------------------------
// Kernel 3: fused expert tile. grid = (T/16 token tiles, E+1 experts),
// 128 threads = 4 waves. Expert EEXP == shared expert (gate weight 1).
// All matrix operands are pre-converted bf16 in workspace -> the inner loops
// are pure {ds_load_b128 / global_load_b128 / v_wmma}.
// ---------------------------------------------------------------------------
__global__ void __launch_bounds__(128)
moe_expert_tile(const unsigned short* __restrict__ xb,    // [T,D] bf16
                const unsigned short* __restrict__ Wgb,   // [E,H,D] bf16
                const unsigned short* __restrict__ Wvb,   // [E,H,D] bf16
                const unsigned short* __restrict__ Wob,   // [E,D,H] bf16
                const unsigned short* __restrict__ Wgsb,  // [H,D] bf16
                const unsigned short* __restrict__ Wvsb,  // [H,D] bf16
                const unsigned short* __restrict__ Wosb,  // [D,H] bf16
                const float* __restrict__ gws,
                float* __restrict__ out)
{
    extern __shared__ unsigned int smem[];
    unsigned int* sx = smem;            // 16 x XS  (x tile, bf16x2)
    unsigned int* sh = smem + 16 * XS;  // 16 x HS  (h tile, bf16x2)
    __shared__ float wgt[16];
    __shared__ int   anyflag;

    const int tile = blockIdx.x;
    const int e    = blockIdx.y;
    const int tid  = threadIdx.x;
    const int lane = tid & 31;
    const int w    = tid >> 5;
    const int t0   = tile * 16;

    if (tid == 0) anyflag = 0;
    __syncthreads();
    if (tid < 16) {
        float g = (e < EEXP) ? gws[(t0 + tid) * EEXP + e] : 1.0f;
        wgt[tid] = g;
        if (g != 0.f) atomicOr(&anyflag, 1);
    }
    __syncthreads();
    if (!anyflag) return;   // no token in this tile routes to this expert

    // ---- stage 0: stage bf16 x tile (16 x 1024) into LDS via TDM ----
    tdm_load_x_tile(xb + (size_t)t0 * DDIM, sx, tid);
    __syncthreads();

    const unsigned short* Wg_e = (e < EEXP) ? (Wgb + (size_t)e * HDIM * DDIM) : Wgsb;
    const unsigned short* Wv_e = (e < EEXP) ? (Wvb + (size_t)e * HDIM * DDIM) : Wvsb;
    const unsigned short* Wo_e = (e < EEXP) ? (Wob + (size_t)e * DDIM * HDIM) : Wosb;

    const int mlo = lane & 15;
    const int hig = lane >> 4;      // lane half
    const int kbA = hig << 4;       // A operand: K+0 / K+16 (contiguous, matches B)
    const int kbB = hig << 4;       // B operand: K+0 / K+16

    unsigned short* shh = (unsigned short*)sh;  // bf16 view of h tile

    // ---- stage 1: h[16,H] = silu(x Wg^T) * (x Wv^T) * gate ----
    for (int n0 = w * 16; n0 < HDIM; n0 += 64) {
        v8f accg = {};
        v8f accv = {};
        const unsigned short* gRow = Wg_e + (size_t)(n0 + mlo) * DDIM + kbB;
        const unsigned short* vRow = Wv_e + (size_t)(n0 + mlo) * DDIM + kbB;
#pragma unroll 2
        for (int k0 = 0; k0 < DDIM; k0 += 32) {
            BOp a, bg, bv;
            const uint4* pa0 = (const uint4*)&sx[mlo * XS + ((k0 + kbA) >> 1)];
            const uint4* pa1 = (const uint4*)&sx[mlo * XS + ((k0 + kbA + 8) >> 1)];
            uint4 a0 = *pa0, a1 = *pa1;
            a.u[0] = a0.x; a.u[1] = a0.y; a.u[2] = a0.z; a.u[3] = a0.w;
            a.u[4] = a1.x; a.u[5] = a1.y; a.u[6] = a1.z; a.u[7] = a1.w;

            const uint4* pg = (const uint4*)(gRow + k0);
            const uint4* pv = (const uint4*)(vRow + k0);
            __builtin_prefetch(gRow + k0 + 64, 0, 1);
            __builtin_prefetch(vRow + k0 + 64, 0, 1);
            uint4 g0 = pg[0], g1 = pg[1];
            uint4 v0 = pv[0], v1 = pv[1];
            bg.u[0] = g0.x; bg.u[1] = g0.y; bg.u[2] = g0.z; bg.u[3] = g0.w;
            bg.u[4] = g1.x; bg.u[5] = g1.y; bg.u[6] = g1.z; bg.u[7] = g1.w;
            bv.u[0] = v0.x; bv.u[1] = v0.y; bv.u[2] = v0.z; bv.u[3] = v0.w;
            bv.u[4] = v1.x; bv.u[5] = v1.y; bv.u[6] = v1.z; bv.u[7] = v1.w;

            accg = __builtin_amdgcn_wmma_f32_16x16x32_bf16(
                false, a.v, false, bg.v, (short)0, accg, false, false);
            accv = __builtin_amdgcn_wmma_f32_16x16x32_bf16(
                false, a.v, false, bv.v, (short)0, accv, false, false);
        }
        int n = n0 + mlo;
#pragma unroll
        for (int r = 0; r < 8; ++r) {
            int m = (hig << 3) + r;                 // token row per C layout
            float g = accg[r];
            float sig = __builtin_amdgcn_rcpf(1.0f + __expf(-g));
            float hv = g * sig * accv[r] * wgt[m];
            shh[m * (HS * 2) + n] = bf16_1(hv);
        }
    }
    __syncthreads();

    // ---- stage 2: out[16,D] += h @ Wo^T ----
    for (int n0 = w * 16; n0 < DDIM; n0 += 64) {
        v8f acc = {};
        const unsigned short* oRow = Wo_e + (size_t)(n0 + mlo) * HDIM + kbB;
#pragma unroll 4
        for (int k0 = 0; k0 < HDIM; k0 += 32) {
            BOp a, bb;
            const uint4* pa0 = (const uint4*)&sh[mlo * HS + ((k0 + kbA) >> 1)];
            const uint4* pa1 = (const uint4*)&sh[mlo * HS + ((k0 + kbA + 8) >> 1)];
            uint4 a0 = *pa0, a1 = *pa1;
            a.u[0] = a0.x; a.u[1] = a0.y; a.u[2] = a0.z; a.u[3] = a0.w;
            a.u[4] = a1.x; a.u[5] = a1.y; a.u[6] = a1.z; a.u[7] = a1.w;

            const uint4* po = (const uint4*)(oRow + k0);
            __builtin_prefetch(oRow + k0 + 64, 0, 1);
            uint4 o0 = po[0], o1 = po[1];
            bb.u[0] = o0.x; bb.u[1] = o0.y; bb.u[2] = o0.z; bb.u[3] = o0.w;
            bb.u[4] = o1.x; bb.u[5] = o1.y; bb.u[6] = o1.z; bb.u[7] = o1.w;

            acc = __builtin_amdgcn_wmma_f32_16x16x32_bf16(
                false, a.v, false, bb.v, (short)0, acc, false, false);
        }
        int d = n0 + mlo;
#pragma unroll
        for (int r = 0; r < 8; ++r) {
            int m = (hig << 3) + r;
            atomicAdd(&out[(size_t)(t0 + m) * DDIM + d], acc[r]);
        }
    }
}

// ---------------------------------------------------------------------------
extern "C" void kernel_launch(void* const* d_in, const int* in_sizes, int n_in,
                              void* d_out, int out_size, void* d_ws, size_t ws_size,
                              hipStream_t stream) {
    const float* x    = (const float*)d_in[0];
    const float* Wr   = (const float*)d_in[1];
    const float* br   = (const float*)d_in[2];
    const float* Wg   = (const float*)d_in[3];
    const float* Wv   = (const float*)d_in[4];
    const float* Wo   = (const float*)d_in[5];
    const float* bo   = (const float*)d_in[6];
    const float* Wg_s = (const float*)d_in[7];
    const float* Wv_s = (const float*)d_in[8];
    const float* Wo_s = (const float*)d_in[9];
    const float* bo_s = (const float*)d_in[10];

    float* out        = (float*)d_out;                       // [T,D]
    float* gate_out   = out + (size_t)TTOK * DDIM;           // [T,E]
    float* topk_out   = gate_out + (size_t)TTOK * EEXP;      // [T,K]

    // ---- workspace layout ----
    char* ws = (char*)d_ws;
    float* gws = (float*)ws;                                  // [T,E] f32
    size_t off = ((size_t)TTOK * EEXP * sizeof(float) + 255) & ~(size_t)255;
    unsigned short* xb   = (unsigned short*)(ws + off); off += (size_t)TTOK * DDIM * 2;
    unsigned short* Wgb  = (unsigned short*)(ws + off); off += (size_t)EEXP * HDIM * DDIM * 2;
    unsigned short* Wvb  = (unsigned short*)(ws + off); off += (size_t)EEXP * HDIM * DDIM * 2;
    unsigned short* Wob  = (unsigned short*)(ws + off); off += (size_t)EEXP * DDIM * HDIM * 2;
    unsigned short* Wgsb = (unsigned short*)(ws + off); off += (size_t)HDIM * DDIM * 2;
    unsigned short* Wvsb = (unsigned short*)(ws + off); off += (size_t)HDIM * DDIM * 2;
    unsigned short* Wosb = (unsigned short*)(ws + off); off += (size_t)DDIM * HDIM * 2;

    // ---- 0) bf16 conversion prepass ----
    auto cvt = [&](const float* s, unsigned short* d, long nelem) {
        long np = nelem >> 1;
        cvt_f32_bf16<<<(unsigned)((np + 255) / 256), 256, 0, stream>>>(
            s, (unsigned int*)d, np);
    };
    cvt(x,    xb,   (long)TTOK * DDIM);
    cvt(Wg,   Wgb,  (long)EEXP * HDIM * DDIM);
    cvt(Wv,   Wvb,  (long)EEXP * HDIM * DDIM);
    cvt(Wo,   Wob,  (long)EEXP * DDIM * HDIM);
    cvt(Wg_s, Wgsb, (long)HDIM * DDIM);
    cvt(Wv_s, Wvsb, (long)HDIM * DDIM);
    cvt(Wo_s, Wosb, (long)DDIM * HDIM);

    // ---- 1) router ----
    moe_router<<<TTOK / 8, 256, 0, stream>>>(x, Wr, br, gate_out, topk_out, gws);

    // ---- 2) bias / zero init of out ----
    moe_bias_init<<<(TTOK * DDIM) / 256, 256, 0, stream>>>(bo, bo_s, gws, out);

    // ---- 3) fused experts (8 routed + 1 shared), top-2 gate folded into h ----
    size_t shbytes = (size_t)(16 * XS + 16 * HS) * sizeof(unsigned int);
    static bool attr_set = false;
    if (!attr_set) {
        hipFuncSetAttribute((const void*)moe_expert_tile,
                            hipFuncAttributeMaxDynamicSharedMemorySize,
                            (int)shbytes);
        attr_set = true;
    }
    dim3 grid(TTOK / 16, EEXP + 1);
    moe_expert_tile<<<grid, 128, shbytes, stream>>>(
        xb, Wgb, Wvb, Wob, Wgsb, Wvsb, Wosb, gws, out);
}
